// SRNN_28948079575465
// MI455X (gfx1250) — matrix operand order repaired and checked by reading, not compiled, not run on hardware
//
#include <hip/hip_runtime.h>
#include <hip/hip_bf16.h>
#include <stdint.h>

// ---------------------------------------------------------------------------
// SRNN (social RNN) forward for MI455X / gfx1250.
// Per-timestep fused pipeline; all matmuls via v_wmma_f32_16x16x32_bf16 with
// pre-packed bf16 fragment-layout operands. B panels are staged block-wide
// into LDS with gfx1250 async loads (GLOBAL_LOAD_ASYNC_TO_LDS_B128 +
// s_wait_asynccnt), eliminating the 8x redundant B loads across waves.
// ---------------------------------------------------------------------------

#define S_LEN 50
#define NENV  128
#define HNUM  32
#define EMB   64
#define ERNN  256
#define NRNN  128
#define ATT   64
#define OUTD  256

#define KCMAX 16   // max K/32 over all GEMMs (K=512 for edge_embed)

typedef __attribute__((ext_vector_type(16))) __bf16 v16bf;
typedef __attribute__((ext_vector_type(8)))  float  v8f;

static __device__ inline unsigned short f2bf_u16(float x) {
    unsigned u = __float_as_uint(x);
    u += 0x7FFFu + ((u >> 16) & 1u);          // round-to-nearest-even
    return (unsigned short)(u >> 16);
}

// ---------------------------------------------------------------------------
// Pack A[M,K] f32 (optionally row-scaled by mask[m/maskDiv]) into WMMA
// A-fragment order: [M/16][K/32][lane 0..31][16 bf16], one contiguous 32B
// chunk per lane. A-fragment element j of lane l (row m = l%16, half hf=l/16):
//   k = kc*32 + hf*16 + j
// ---------------------------------------------------------------------------
__global__ __launch_bounds__(256) void pack_a_bf16(
    unsigned short* __restrict__ Ap, const float* __restrict__ A,
    const float* __restrict__ mask, int maskDiv, int M, int K)
{
    int idx = blockIdx.x * 256 + threadIdx.x;
    if (idx >= M * K) return;
    int frag = idx >> 9;          // / 512
    int r    = idx & 511;
    int lane = r >> 4, j = r & 15;
    int KC = K >> 5;
    int mt = frag / KC, kc = frag - mt * KC;
    int m  = mt * 16 + (lane & 15);
    int hf = lane >> 4;
    int k  = kc * 32 + hf * 16 + j;
    float v = A[(size_t)m * K + k];
    if (mask) v *= mask[m / maskDiv];
    Ap[idx] = f2bf_u16(v);
}

// ---------------------------------------------------------------------------
// Pack W[N,K] f32 into WMMA B-fragment order (B[k][n] = W[n][k]):
// [N/16][K/32][lane 0..31][16 bf16]; lane l holds col n = l%16,
// K-range kc*32 + (l/16)*16 + [0..15].
// ---------------------------------------------------------------------------
__global__ __launch_bounds__(256) void pack_w_bf16(
    unsigned short* __restrict__ Wp, const float* __restrict__ W, int N, int K)
{
    int idx = blockIdx.x * 256 + threadIdx.x;
    if (idx >= N * K) return;
    int frag = idx >> 9;
    int r    = idx & 511;
    int lane = r >> 4, j = r & 15;
    int KC = K >> 5;
    int nt = frag / KC, kc = frag - nt * KC;
    int n  = nt * 16 + (lane & 15);
    int hf = lane >> 4;
    int k  = kc * 32 + hf * 16 + j;
    Wp[idx] = f2bf_u16(W[(size_t)n * K + k]);
}

// ---------------------------------------------------------------------------
// GEMM on packed operands: C[M,N] = act( A @ W^T + bias ).
// Block = 256 threads = 8 waves; wave w owns a 16x32 tile (rows
// blockIdx.y*128 + w*16, cols blockIdx.x*32 .. +31), two accumulators share
// one A fragment. The block's B panel (2 n-tiles x all K-chunks, contiguous
// in packed layout) is staged once into LDS via async loads; inner loop is
// 1 global 32B A-load + 2 LDS 32B B-loads + 2 WMMAs.
// Requires M % 128 == 0, N % 32 == 0, K % 32 == 0, K/32 even and <= KCMAX.
// ---------------------------------------------------------------------------
__global__ __launch_bounds__(256) void gemm_wmma_packed(
    float* __restrict__ C,
    const unsigned short* __restrict__ Ap,
    const unsigned short* __restrict__ Wp,
    const float* __restrict__ bias,
    int M, int N, int K, int relu)
{
    __shared__ __attribute__((aligned(32))) unsigned short Bsh[2 * KCMAX * 512];

    const int lane = threadIdx.x & 31;
    const int wave = threadIdx.x >> 5;
    const int n0   = blockIdx.x * 32;
    const int m0   = blockIdx.y * 128 + wave * 16;
    const int KC   = K >> 5;

    // ---- stage B panel (contiguous 2*KC*512 ushorts) into LDS, async ----
    {
        const unsigned short* __restrict__ src = Wp + (size_t)(n0 >> 4) * KC * 512;
        const int nchunk = 2 * KC * 64;            // 16-byte chunks
        for (int c = threadIdx.x; c < nchunk; c += 256) {
            unsigned lds_addr = (unsigned)(uintptr_t)(&Bsh[(size_t)c * 8]);
            const unsigned short* g = src + (size_t)c * 8;
            asm volatile("global_load_async_to_lds_b128 %0, %1, off"
                         :: "v"(lds_addr), "v"(g)
                         : "memory");
        }
        asm volatile("s_wait_asynccnt 0" ::: "memory");
        __syncthreads();
    }

    const v16bf* __restrict__ ap   = (const v16bf*)Ap + (size_t)(m0 >> 4) * KC * 32 + lane;
    const v16bf* __restrict__ bsh0 = (const v16bf*)Bsh + lane;           // tile 0 frags
    const v16bf* __restrict__ bsh1 = (const v16bf*)Bsh + KC * 32 + lane; // tile 1 frags

    v8f acc0 = {}, acc1 = {};
    for (int kc = 0; kc < KC; ++kc) {
        v16bf af = ap[(size_t)kc * 32];
        v16bf b0 = bsh0[kc * 32];
        v16bf b1 = bsh1[kc * 32];
        acc0 = __builtin_amdgcn_wmma_f32_16x16x32_bf16(
            false, af, false, b0, (short)0, acc0, false, false);
        acc1 = __builtin_amdgcn_wmma_f32_16x16x32_bf16(
            false, af, false, b1, (short)0, acc1, false, false);
    }

    const int hf  = lane >> 4;
    const int nc0 = n0 + (lane & 15);
    const int nc1 = nc0 + 16;
    const float bv0 = bias ? bias[nc0] : 0.0f;
    const float bv1 = bias ? bias[nc1] : 0.0f;
    // C/D layout: VGPR r, lanes 0-15 -> row m0+r, lanes 16-31 -> row m0+r+8.
    #pragma unroll
    for (int r = 0; r < 8; ++r) {
        const int row = m0 + r + hf * 8;
        float v0 = acc0[r] + bv0;
        float v1 = acc1[r] + bv1;
        if (relu) { v0 = fmaxf(v0, 0.0f); v1 = fmaxf(v1, 0.0f); }
        C[(size_t)row * N + nc0] = v0;
        C[(size_t)row * N + nc1] = v1;
    }
}

// ---------------------------------------------------------------------------
// Small encoder: out[M,64] = relu(in[M,din] @ w[64,din]^T + b)   (din = 2 or 7)
// ---------------------------------------------------------------------------
__global__ __launch_bounds__(256) void enc_linear_relu(
    float* __restrict__ out, const float* __restrict__ in,
    const float* __restrict__ w, const float* __restrict__ b,
    int M, int din)
{
    int idx = blockIdx.x * 256 + threadIdx.x;
    if (idx >= M * 64) return;
    int m = idx >> 6, n = idx & 63;
    float acc = b[n];
    for (int k = 0; k < din; ++k) acc += in[m * din + k] * w[n * din + k];
    out[idx] = fmaxf(acc, 0.0f);
}

// ---------------------------------------------------------------------------
// GRU gate combine: h <- (1-z)*n + z*(h*mask); gi/gh are [M,3*dh] (r,z,n).
// ---------------------------------------------------------------------------
__global__ __launch_bounds__(256) void gru_combine(
    float* __restrict__ h,
    const float* __restrict__ gi, const float* __restrict__ gh,
    const float* __restrict__ mask, int maskDiv,
    int M, int dh)
{
    int idx = blockIdx.x * 256 + threadIdx.x;
    if (idx >= M * dh) return;
    int m = idx / dh, j = idx - m * dh;
    size_t base = (size_t)m * 3 * dh;
    float hm = h[idx] * mask[m / maskDiv];
    float ir = gi[base + j],          hr = gh[base + j];
    float iz = gi[base + dh + j],     hz = gh[base + dh + j];
    float in = gi[base + 2 * dh + j], hn = gh[base + 2 * dh + j];
    float r = 1.0f / (1.0f + __expf(-(ir + hr)));
    float z = 1.0f / (1.0f + __expf(-(iz + hz)));
    float n = tanhf(in + r * hn);
    h[idx] = (1.0f - z) * n + z * hm;
}

// ---------------------------------------------------------------------------
// Attention for one step: one block per env. scores = (te.se_h)*4; softmax(H=32).
// ---------------------------------------------------------------------------
__global__ __launch_bounds__(256) void attention_step(
    float* __restrict__ wgt,
    const float* __restrict__ te, const float* __restrict__ se,
    const float* __restrict__ hs)
{
    __shared__ float s_te[ATT];
    __shared__ float s_at[HNUM];
    const int nb  = blockIdx.x;
    const int tid = threadIdx.x;

    if (tid < ATT) s_te[tid] = te[nb * ATT + tid];
    __syncthreads();

    if (tid < HNUM) {
        const float* sp = se + (size_t)(nb * HNUM + tid) * ATT;
        float sc = 0.0f;
        #pragma unroll
        for (int k = 0; k < ATT; ++k) sc += s_te[k] * sp[k];
        s_at[tid] = sc * 4.0f;   // H/sqrt(ATT) = 32/8
    }
    __syncthreads();

    if (tid == 0) {
        float mx = -1e30f;
        for (int h = 0; h < HNUM; ++h) mx = fmaxf(mx, s_at[h]);
        float sum = 0.0f;
        for (int h = 0; h < HNUM; ++h) { float e = __expf(s_at[h] - mx); s_at[h] = e; sum += e; }
        float inv = 1.0f / sum;
        for (int h = 0; h < HNUM; ++h) s_at[h] *= inv;
    }
    __syncthreads();

    float acc = 0.0f;
    for (int h = 0; h < HNUM; ++h)
        acc += s_at[h] * hs[(size_t)(nb * HNUM + h) * ERNN + tid];
    wgt[nb * ERNN + tid] = acc;
}

// ---------------------------------------------------------------------------
// concat along features: out[M, da+db] = [a[M,da] | b[M,db]]
// ---------------------------------------------------------------------------
__global__ __launch_bounds__(256) void concat2(
    float* __restrict__ out, const float* __restrict__ a,
    const float* __restrict__ b, int da, int db, int M)
{
    int tot = da + db;
    int idx = blockIdx.x * 256 + threadIdx.x;
    if (idx >= M * tot) return;
    int m = idx / tot, k = idx - m * tot;
    out[idx] = (k < da) ? a[m * da + k] : b[m * db + (k - da)];
}

// ---------------------------------------------------------------------------
// initial hidden-state unpack.
// ---------------------------------------------------------------------------
__global__ __launch_bounds__(256) void init_state(
    float* __restrict__ h_t, float* __restrict__ h_s, float* __restrict__ h_n,
    const float* __restrict__ h_edge, const float* __restrict__ h_node)
{
    int idx = blockIdx.x * 256 + threadIdx.x;
    if (idx < NENV * ERNN) {
        int n = idx >> 8, d = idx & 255;
        h_t[idx] = h_edge[((size_t)n * (1 + HNUM) + 0) * ERNN + d];
    }
    if (idx < NENV * HNUM * ERNN) {
        int r = idx >> 8, d = idx & 255;
        int n = r >> 5, h = r & 31;
        h_s[idx] = h_edge[((size_t)n * (1 + HNUM) + 1 + h) * ERNN + d];
    }
    if (idx < NENV * NRNN) h_n[idx] = h_node[idx];
}

// ---------------------------------------------------------------------------
// host launcher
// ---------------------------------------------------------------------------
static inline void gemm_packed(float* C, const float* A, const unsigned short* Wp,
                               const float* bias, const float* mask, int maskDiv,
                               int M, int N, int K, int relu,
                               unsigned short* Ap, hipStream_t s)
{
    pack_a_bf16<<<(M * K + 255) / 256, 256, 0, s>>>(Ap, A, mask, maskDiv, M, K);
    dim3 g(N / 32, M / 128);
    gemm_wmma_packed<<<g, 256, 0, s>>>(C, Ap, Wp, bias, M, N, K, relu);
}

extern "C" void kernel_launch(void* const* d_in, const int* in_sizes, int n_in,
                              void* d_out, int out_size, void* d_ws, size_t ws_size,
                              hipStream_t stream)
{
    // ---- inputs (setup_inputs dict order; params tuples flattened in order) ----
    const float* robot_node     = (const float*)d_in[0];  // [S,N,1,7]
    const float* temporal_edges = (const float*)d_in[1];  // [S,N,1,2]
    const float* spatial_edges  = (const float*)d_in[2];  // [S,N,H,2]
    const float* masks          = (const float*)d_in[3];  // [S,N]
    const float* h_node_in      = (const float*)d_in[4];  // [N,1,128]
    const float* h_edge_in      = (const float*)d_in[5];  // [N,33,256]
    const float* t_enc_w  = (const float*)d_in[6];   const float* t_enc_b  = (const float*)d_in[7];
    const float* t_wi     = (const float*)d_in[8];   const float* t_wh     = (const float*)d_in[9];
    const float* t_bi     = (const float*)d_in[10];  const float* t_bh     = (const float*)d_in[11];
    const float* s_enc_w  = (const float*)d_in[12];  const float* s_enc_b  = (const float*)d_in[13];
    const float* s_wi     = (const float*)d_in[14];  const float* s_wh     = (const float*)d_in[15];
    const float* s_bi     = (const float*)d_in[16];  const float* s_bh     = (const float*)d_in[17];
    const float* at_w     = (const float*)d_in[18];  const float* at_b     = (const float*)d_in[19];
    const float* as_w     = (const float*)d_in[20];  const float* as_b     = (const float*)d_in[21];
    const float* n_enc_w  = (const float*)d_in[22];  const float* n_enc_b  = (const float*)d_in[23];
    const float* ee_w     = (const float*)d_in[24];  const float* ee_b     = (const float*)d_in[25];
    const float* n_wi     = (const float*)d_in[26];  const float* n_wh     = (const float*)d_in[27];
    const float* n_bi     = (const float*)d_in[28];  const float* n_bh     = (const float*)d_in[29];
    const float* no_w     = (const float*)d_in[30];  const float* no_b     = (const float*)d_in[31];

    float* out = (float*)d_out;   // [S,N,1,256]

    // ---- workspace layout (64B-aligned chunks) ----
    float* Wbase = (float*)d_ws;
    size_t off = 0;
    auto allocf = [&](size_t n) {
        n = (n + 15) & ~(size_t)15;
        float* p = Wbase + off; off += n; return p;
    };
    auto allocu = [&](size_t n) { return (unsigned short*)allocf((n + 1) / 2); };

    float* h_t   = allocf(NENV * ERNN);
    float* h_s   = allocf(NENV * HNUM * ERNN);
    float* h_n   = allocf(NENV * NRNN);
    float* t_in  = allocf(NENV * EMB);
    float* s_in  = allocf(NENV * HNUM * EMB);
    float* pos   = allocf(NENV * EMB);
    float* gi_t  = allocf(NENV * 3 * ERNN);
    float* gh_t  = allocf(NENV * 3 * ERNN);
    float* gi_s  = allocf(NENV * HNUM * 3 * ERNN);
    float* gh_s  = allocf(NENV * HNUM * 3 * ERNN);
    float* te    = allocf(NENV * ATT);
    float* se    = allocf(NENV * HNUM * ATT);
    float* wgt   = allocf(NENV * ERNN);
    float* cat5  = allocf(NENV * 2 * ERNN);
    float* eemb  = allocf(NENV * EMB);
    float* nin   = allocf(NENV * 2 * EMB);
    float* gi_n  = allocf(NENV * 3 * NRNN);
    float* gh_n  = allocf(NENV * 3 * NRNN);

    // packed bf16 weights (converted once per call, reused 50 steps x N/16 tiles)
    unsigned short* Wp_twi = allocu(3 * ERNN * EMB);        // [768,64]
    unsigned short* Wp_twh = allocu(3 * ERNN * ERNN);       // [768,256]
    unsigned short* Wp_swi = allocu(3 * ERNN * EMB);        // [768,64]
    unsigned short* Wp_swh = allocu(3 * ERNN * ERNN);       // [768,256]
    unsigned short* Wp_at  = allocu(ATT * ERNN);            // [64,256]
    unsigned short* Wp_as  = allocu(ATT * ERNN);            // [64,256]
    unsigned short* Wp_ee  = allocu(EMB * 2 * ERNN);        // [64,512]
    unsigned short* Wp_nwi = allocu(3 * NRNN * 2 * EMB);    // [384,128]
    unsigned short* Wp_nwh = allocu(3 * NRNN * NRNN);       // [384,128]
    unsigned short* Wp_no  = allocu(OUTD * NRNN);           // [256,128]
    // shared packed-A scratch, sized for the largest A (4096 x 256)
    unsigned short* Ap     = allocu((size_t)NENV * HNUM * ERNN);
    (void)ws_size;

    const int MS = NENV * HNUM;   // 4096 spatial rows

    // ---- one-time (per call) weight packing ----
    auto packw = [&](unsigned short* dst, const float* w, int N, int K) {
        pack_w_bf16<<<(N * K + 255) / 256, 256, 0, stream>>>(dst, w, N, K);
    };
    packw(Wp_twi, t_wi, 3 * ERNN, EMB);
    packw(Wp_twh, t_wh, 3 * ERNN, ERNN);
    packw(Wp_swi, s_wi, 3 * ERNN, EMB);
    packw(Wp_swh, s_wh, 3 * ERNN, ERNN);
    packw(Wp_at,  at_w, ATT, ERNN);
    packw(Wp_as,  as_w, ATT, ERNN);
    packw(Wp_ee,  ee_w, EMB, 2 * ERNN);
    packw(Wp_nwi, n_wi, 3 * NRNN, 2 * EMB);
    packw(Wp_nwh, n_wh, 3 * NRNN, NRNN);
    packw(Wp_no,  no_w, OUTD, NRNN);

    init_state<<<(MS * ERNN + 255) / 256, 256, 0, stream>>>(h_t, h_s, h_n, h_edge_in, h_node_in);

    for (int t = 0; t < S_LEN; ++t) {
        const float* mk = masks + (size_t)t * NENV;

        // ---- encoders ----
        enc_linear_relu<<<(NENV * 64 + 255) / 256, 256, 0, stream>>>(
            t_in, temporal_edges + (size_t)t * NENV * 2, t_enc_w, t_enc_b, NENV, 2);
        enc_linear_relu<<<(MS * 64 + 255) / 256, 256, 0, stream>>>(
            s_in, spatial_edges + (size_t)t * MS * 2, s_enc_w, s_enc_b, MS, 2);
        enc_linear_relu<<<(NENV * 64 + 255) / 256, 256, 0, stream>>>(
            pos, robot_node + (size_t)t * NENV * 7, n_enc_w, n_enc_b, NENV, 7);

        // ---- temporal edge GRU step ----
        gemm_packed(gi_t, t_in, Wp_twi, t_bi, nullptr, 1, NENV, 3 * ERNN, EMB, 0, Ap, stream);
        gemm_packed(gh_t, h_t,  Wp_twh, t_bh, mk,      1, NENV, 3 * ERNN, ERNN, 0, Ap, stream);
        gru_combine<<<(NENV * ERNN + 255) / 256, 256, 0, stream>>>(
            h_t, gi_t, gh_t, mk, 1, NENV, ERNN);

        // ---- spatial edge GRU step (dominant GEMMs) ----
        gemm_packed(gi_s, s_in, Wp_swi, s_bi, nullptr, 1,    MS, 3 * ERNN, EMB, 0, Ap, stream);
        gemm_packed(gh_s, h_s,  Wp_swh, s_bh, mk,      HNUM, MS, 3 * ERNN, ERNN, 0, Ap, stream);
        gru_combine<<<(MS * ERNN + 255) / 256, 256, 0, stream>>>(
            h_s, gi_s, gh_s, mk, HNUM, MS, ERNN);

        // ---- attention ----
        gemm_packed(te, h_t, Wp_at, at_b, nullptr, 1, NENV, ATT, ERNN, 0, Ap, stream);
        gemm_packed(se, h_s, Wp_as, as_b, nullptr, 1, MS,   ATT, ERNN, 0, Ap, stream);
        attention_step<<<NENV, 256, 0, stream>>>(wgt, te, se, h_s);

        // ---- node input path ----
        concat2<<<(NENV * 2 * ERNN + 255) / 256, 256, 0, stream>>>(
            cat5, h_t, wgt, ERNN, ERNN, NENV);
        gemm_packed(eemb, cat5, Wp_ee, ee_b, nullptr, 1, NENV, EMB, 2 * ERNN, 1, Ap, stream);
        concat2<<<(NENV * 2 * EMB + 255) / 256, 256, 0, stream>>>(
            nin, pos, eemb, EMB, EMB, NENV);

        // ---- node GRU step ----
        gemm_packed(gi_n, nin, Wp_nwi, n_bi, nullptr, 1, NENV, 3 * NRNN, 2 * EMB, 0, Ap, stream);
        gemm_packed(gh_n, h_n, Wp_nwh, n_bh, mk,      1, NENV, 3 * NRNN, NRNN, 0, Ap, stream);
        gru_combine<<<(NENV * NRNN + 255) / 256, 256, 0, stream>>>(
            h_n, gi_n, gh_n, mk, 1, NENV, NRNN);

        // ---- output projection for this step ----
        gemm_packed(out + (size_t)t * NENV * OUTD, h_n, Wp_no, no_b, nullptr, 1,
                    NENV, OUTD, NRNN, 0, Ap, stream);
    }
}
